// AttentionProjector_40140764348434
// MI455X (gfx1250) — matrix operand order repaired and checked by reading, not compiled, not run on hardware
//
#include <hip/hip_runtime.h>
#include <hip/hip_bf16.h>

// ---------------------------------------------------------------------------
// AttentionProjector for MI455X (gfx1250, wave32, WMMA).
//   keys   = x @ Wk^T + bk         [8192 x 4096]   (bf16, row-major = B^T of QK^T)
//   valsT  = (x @ Wv^T + bv)^T     [4096 x 8192]   (bf16, = B^T of scores@V)
//   scores = softmax(q @ keys^T)   [256 x 8192]    (fp32)
//   out    = scores @ vals         [256 x 4096]    (fp32)
// All GEMMs use v_wmma_f32_16x16x32_bf16 with fp32 accumulation.
// Double-buffered LDS: one barrier per K-tile; next tile's global loads are
// issued a full compute phase ahead. fp32 sources convert to bf16 with the
// native cast (v_cvt*bf16) while staging into LDS.
// ---------------------------------------------------------------------------

#define BM 128
#define BN 128
#define BK 64
#define LDT 80   // padded LDS row stride (elements); 160B keeps 16B alignment

typedef __attribute__((ext_vector_type(16))) __bf16 v16bf;
typedef __attribute__((ext_vector_type(8)))  __bf16 v8bf;
typedef __attribute__((ext_vector_type(8)))  float  v8f;

// Per-thread staging registers: 32 contiguous source elements.
template <typename T> struct Stage;

template <> struct Stage<float> {
  float4 r[8];
  __device__ __forceinline__ void load(const float* __restrict__ src) {
    const float4* s4 = (const float4*)src;
#pragma unroll
    for (int j = 0; j < 8; ++j) r[j] = s4[j];
  }
  __device__ __forceinline__ void store(__bf16* __restrict__ dst) const {
#pragma unroll
    for (int j = 0; j < 4; ++j) {
      float4 a = r[2 * j], b = r[2 * j + 1];
      v8bf o;
      o[0] = (__bf16)a.x; o[1] = (__bf16)a.y;
      o[2] = (__bf16)a.z; o[3] = (__bf16)a.w;
      o[4] = (__bf16)b.x; o[5] = (__bf16)b.y;
      o[6] = (__bf16)b.z; o[7] = (__bf16)b.w;
      *(v8bf*)(dst + 8 * j) = o;
    }
  }
};

template <> struct Stage<__bf16> {
  v8bf r[4];
  __device__ __forceinline__ void load(const __bf16* __restrict__ src) {
#pragma unroll
    for (int j = 0; j < 4; ++j) r[j] = ((const v8bf*)src)[j];
  }
  __device__ __forceinline__ void store(__bf16* __restrict__ dst) const {
#pragma unroll
    for (int j = 0; j < 4; ++j) *(v8bf*)(dst + 8 * j) = r[j];
  }
};

// Two aligned 16B LDS reads -> one 16-element bf16 fragment.
__device__ __forceinline__ v16bf frag16(const __bf16* p0, const __bf16* p1) {
  v8bf lo = *(const v8bf*)p0;
  v8bf hi = *(const v8bf*)p1;
  return __builtin_shufflevector(lo, hi, 0, 1, 2, 3, 4, 5, 6, 7,
                                         8, 9, 10, 11, 12, 13, 14, 15);
}

// C = A[M x K] (fp32, row-major) * Bt[Ncols x K]^T (+bias), tiles 128x128x64.
// EPI: 0 -> fp32 out [M x Ncols]; 1 -> bf16 out [M x Ncols];
//      2 -> bf16 out transposed [Ncols x M] (ldOut = M stride).
template <typename TB, int EPI>
__global__ __launch_bounds__(256) void gemm_bf16_wmma(
    const float* __restrict__ A, const TB* __restrict__ Bt,
    const float* __restrict__ bias, void* __restrict__ outp,
    int M, int Ncols, int K, int ldOut) {
  __shared__ __align__(16) __bf16 sA[2][BM * LDT];
  __shared__ __align__(16) __bf16 sB[2][BN * LDT];

  const int tid   = threadIdx.x;
  const int lane  = tid & 31;
  const int wave  = tid >> 5;
  const int waveM = wave >> 2;       // 0..1 : 64 rows each
  const int waveN = wave & 3;        // 0..3 : 32 cols each
  const int tileM = blockIdx.y * BM;
  const int tileN = blockIdx.x * BN;

  const int lrow = tid >> 1;         // 0..127 : tile row this thread stages
  const int lcol = (tid & 1) * 32;   // 0 / 32 : tile col chunk

  const int mHalf = lane >> 4;       // 0/1 : lane-group within wave
  const int mIdx  = lane & 15;

  const v8f vzero = {0.f, 0.f, 0.f, 0.f, 0.f, 0.f, 0.f, 0.f};
  v8f acc[4][2];
#pragma unroll
  for (int i = 0; i < 4; ++i)
#pragma unroll
    for (int j = 0; j < 2; ++j) acc[i][j] = vzero;

  const float* aBase = A + (size_t)(tileM + lrow) * K + lcol;
  const TB*    bBase = Bt + (size_t)(tileN + lrow) * K + lcol;

  Stage<float> aS;
  Stage<TB>    bS;
  aS.load(aBase);
  bS.load(bBase);

  int buf = 0;
  for (int k0 = 0; k0 < K; k0 += BK) {
    __bf16* sAb = sA[buf];
    __bf16* sBb = sB[buf];
    aS.store(sAb + lrow * LDT + lcol);   // convert + stage tile k0
    bS.store(sBb + lrow * LDT + lcol);
    __syncthreads();

    if (k0 + BK < K) {                   // issue next tile's loads early
      __builtin_prefetch(aBase + k0 + 2 * BK, 0, 1);
      __builtin_prefetch(bBase + k0 + 2 * BK, 0, 1);
      aS.load(aBase + k0 + BK);
      bS.load(bBase + k0 + BK);
    }

#pragma unroll
    for (int kk = 0; kk < BK; kk += 32) {
      v16bf af[4];
      v16bf bfr[2];
#pragma unroll
      for (int ms = 0; ms < 4; ++ms) {
        const __bf16* base =
            sAb + (waveM * 64 + ms * 16 + mIdx) * LDT + kk + mHalf * 16;
        af[ms] = frag16(base, base + 8);
      }
#pragma unroll
      for (int ns = 0; ns < 2; ++ns) {
        const __bf16* base =
            sBb + (waveN * 32 + ns * 16 + mIdx) * LDT + kk + mHalf * 16;
        bfr[ns] = frag16(base, base + 8);
      }
#pragma unroll
      for (int ms = 0; ms < 4; ++ms)
#pragma unroll
        for (int ns = 0; ns < 2; ++ns)
          acc[ms][ns] = __builtin_amdgcn_wmma_f32_16x16x32_bf16(
              false, af[ms], false, bfr[ns], (short)0, acc[ms][ns],
              false, false);
    }
    buf ^= 1;
  }

  // Epilogue. C/D layout: elem v of lane -> row = 8*mHalf + v, col = mIdx.
#pragma unroll
  for (int ns = 0; ns < 2; ++ns) {
    const int col = tileN + waveN * 32 + ns * 16 + mIdx;
    const float bv = bias ? bias[col] : 0.0f;
#pragma unroll
    for (int ms = 0; ms < 4; ++ms) {
      const int row0 = tileM + waveM * 64 + ms * 16 + mHalf * 8;
      if (EPI == 0) {
        float* o = (float*)outp;
#pragma unroll
        for (int v = 0; v < 8; ++v)
          o[(size_t)(row0 + v) * ldOut + col] = acc[ms][ns][v] + bv;
      } else if (EPI == 1) {
        __bf16* o = (__bf16*)outp;
#pragma unroll
        for (int v = 0; v < 8; ++v)
          o[(size_t)(row0 + v) * ldOut + col] = (__bf16)(acc[ms][ns][v] + bv);
      } else {  // transposed bf16: 8 consecutive rows -> one packed 16B store
        __bf16* o = (__bf16*)outp;
        v8bf pk;
#pragma unroll
        for (int v = 0; v < 8; ++v) pk[v] = (__bf16)(acc[ms][ns][v] + bv);
        *(v8bf*)&o[(size_t)col * ldOut + row0] = pk;
      }
    }
  }
}

// Row softmax in place: one 256-thread block (8 waves) per row.
__global__ __launch_bounds__(256) void softmax_rows(float* __restrict__ s,
                                                    int ncols) {
  float* p = s + (size_t)blockIdx.x * ncols;
  const int t = threadIdx.x, lane = t & 31, wave = t >> 5;
  __shared__ float redm[8];
  __shared__ float reds[8];

  float m = -3.4e38f;
  for (int c = t; c < ncols; c += 256) m = fmaxf(m, p[c]);
#pragma unroll
  for (int off = 16; off; off >>= 1) m = fmaxf(m, __shfl_xor(m, off, 32));
  if (lane == 0) redm[wave] = m;
  __syncthreads();
  if (wave == 0) {
    float v = (lane < 8) ? redm[lane] : -3.4e38f;
#pragma unroll
    for (int off = 4; off; off >>= 1) v = fmaxf(v, __shfl_xor(v, off, 32));
    if (lane == 0) redm[0] = v;
  }
  __syncthreads();
  m = redm[0];

  float sum = 0.f;
  for (int c = t; c < ncols; c += 256) {
    float e = __expf(p[c] - m);
    p[c] = e;
    sum += e;
  }
#pragma unroll
  for (int off = 16; off; off >>= 1) sum += __shfl_xor(sum, off, 32);
  if (lane == 0) reds[wave] = sum;
  __syncthreads();
  if (wave == 0) {
    float v = (lane < 8) ? reds[lane] : 0.f;
#pragma unroll
    for (int off = 4; off; off >>= 1) v += __shfl_xor(v, off, 32);
    if (lane == 0) reds[0] = v;
  }
  __syncthreads();
  const float inv = 1.f / reds[0];
  for (int c = t; c < ncols; c += 256) p[c] *= inv;
}

extern "C" void kernel_launch(void* const* d_in, const int* in_sizes, int n_in,
                              void* d_out, int out_size, void* d_ws,
                              size_t ws_size, hipStream_t stream) {
  (void)in_sizes; (void)n_in; (void)out_size; (void)ws_size;
  const int L = 256, D = 4096, N = 8192;

  const float* x   = (const float*)d_in[0];  // [1, N, D]
  const float* q   = (const float*)d_in[1];  // [L, D]
  const float* Wk  = (const float*)d_in[2];  // [D, D]
  const float* Wkb = (const float*)d_in[3];  // [D]
  const float* Wv  = (const float*)d_in[4];  // [D, D]
  const float* Wvb = (const float*)d_in[5];  // [D]
  float* out = (float*)d_out;                // [L, D]

  // Workspace: keys bf16 [N][D] | valuesT bf16 [D][N] | scores f32 [L][N]
  __bf16* keys    = (__bf16*)d_ws;
  __bf16* valuesT = (__bf16*)((char*)d_ws + (size_t)N * D * sizeof(__bf16));
  float*  scores  = (float*)((char*)d_ws + (size_t)2 * N * D * sizeof(__bf16));

  dim3 blk(256);
  // keys = x @ Wk^T + bk    (bf16 row-major [N][D])
  gemm_bf16_wmma<float, 1><<<dim3(D / BN, N / BM), blk, 0, stream>>>(
      x, Wk, Wkb, keys, N, D, D, D);
  // valuesT = (x @ Wv^T + bv)^T  (bf16 [D][N])
  gemm_bf16_wmma<float, 2><<<dim3(D / BN, N / BM), blk, 0, stream>>>(
      x, Wv, Wvb, valuesT, N, D, D, N);
  // scores = q @ keys^T   (fp32 [L][N]); keys IS B^T here
  gemm_bf16_wmma<__bf16, 0><<<dim3(N / BN, L / BM), blk, 0, stream>>>(
      q, keys, nullptr, scores, L, N, D, N);
  // softmax over N per row
  softmax_rows<<<dim3(L), blk, 0, stream>>>(scores, N);
  // out = scores @ values  (fp32 [L][D]); valuesT IS B^T here
  gemm_bf16_wmma<__bf16, 0><<<dim3(D / BN, L / BM), blk, 0, stream>>>(
      scores, valuesT, nullptr, out, L, D, N, D);
}
